// MinimalGPT2Attention_29343216566539
// MI455X (gfx1250) — matrix operand order repaired and checked by reading, not compiled, not run on hardware
//
#include <hip/hip_runtime.h>
#include <hip/hip_bf16.h>
#include <cstdint>

// ---------------------------------------------------------------------------
// GPT-2 attention block for gfx1250 (MI455X): bf16 WMMA everywhere.
// B=2, S=2048, D=1024, H=16, HD=64.
// ---------------------------------------------------------------------------

typedef __bf16 bf16_t;
typedef __bf16 v16bf __attribute__((ext_vector_type(16)));
typedef __bf16 v8bf  __attribute__((ext_vector_type(8)));
typedef float  v8f   __attribute__((ext_vector_type(8)));

#define WMMA_BF16(A_, B_, C_) \
  __builtin_amdgcn_wmma_f32_16x16x32_bf16(false, (A_), false, (B_), (short)0, (C_), false, false)

union FragAB { v16bf v; v8bf h[2]; };

// A operand (16x32, 16-bit): lane = M%16; per-lane K chunks are interleaved:
//   lanes 0-15 : elems 0-7 -> K 0-7,  elems 8-15 -> K 16-23
//   lanes 16-31: elems 0-7 -> K 8-15, elems 8-15 -> K 24-31
// Caller passes p = row_ptr + k0 + half*8.
__device__ __forceinline__ void load_fragA(FragAB& f, const bf16_t* p) {
  f.h[0] = *(const v8bf*)p;
  f.h[1] = *(const v8bf*)(p + 16);
}

// B operand (32x16, 16-bit): lane = N%16; per-lane K chunks interleaved (same as A):
//   lanes 0-15 : K 0-7, 16-23;  lanes 16-31: K 8-15, 24-31
// Caller passes p = rowN_ptr + k0 + half*8.
__device__ __forceinline__ void load_fragB(FragAB& f, const bf16_t* p) {
  f.h[0] = *(const v8bf*)p;
  f.h[1] = *(const v8bf*)(p + 16);
}

// ---------------------------------------------------------------------------
// fp32 -> bf16 convert
// ---------------------------------------------------------------------------
__global__ __launch_bounds__(256) void k_cvt(const float* __restrict__ src,
                                             bf16_t* __restrict__ dst, int n) {
  int i = blockIdx.x * 256 + threadIdx.x;
  if (i < n) dst[i] = (bf16_t)src[i];
}

// src [K][N] fp32 -> dst [N][K] bf16 (transpose+convert)
__global__ __launch_bounds__(256) void k_tcvt(const float* __restrict__ src,
                                              bf16_t* __restrict__ dst, int K, int N) {
  long i = (long)blockIdx.x * 256 + threadIdx.x;
  if (i < (long)K * N) {
    int k = (int)(i / N), n = (int)(i % N);
    dst[(size_t)n * K + k] = (bf16_t)src[i];
  }
}

// ---------------------------------------------------------------------------
// QKV GEMM:  [4096 x 1024] (bf16) x WqkvT [3072 x 1024] (bf16) -> scatter
// into Q [B,H,S,64], K [B,H,S,64], V^T [B,H,64,S]  (all bf16, f32 accumulate)
// Wave tile: 64x64. 8 waves / block.
// ---------------------------------------------------------------------------
__global__ __launch_bounds__(256)
void k_gemm_qkv(const bf16_t* __restrict__ A, const bf16_t* __restrict__ Bt,
                const float* __restrict__ bias,
                bf16_t* __restrict__ Qb, bf16_t* __restrict__ Kb,
                bf16_t* __restrict__ VTb) {
  const int NT = 3072 / 64;  // 48 N-tiles
  int wid  = blockIdx.x * 8 + (threadIdx.x >> 5);
  int lane = threadIdx.x & 31;
  int hlf = lane >> 4, lr = lane & 15;
  int m0 = (wid / NT) * 64;
  int n0 = (wid % NT) * 64;

  v8f z = {0.f, 0.f, 0.f, 0.f, 0.f, 0.f, 0.f, 0.f};
  v8f acc[4][4];
#pragma unroll
  for (int i = 0; i < 4; i++)
#pragma unroll
    for (int j = 0; j < 4; j++) acc[i][j] = z;

  for (int k0 = 0; k0 < 1024; k0 += 32) {
    FragAB a[4];
#pragma unroll
    for (int i = 0; i < 4; i++)
      load_fragA(a[i], A + (size_t)(m0 + i * 16 + lr) * 1024 + k0 + hlf * 8);
#pragma unroll
    for (int j = 0; j < 4; j++) {
      FragAB b;
      load_fragB(b, Bt + (size_t)(n0 + j * 16 + lr) * 1024 + k0 + hlf * 8);
#pragma unroll
      for (int i = 0; i < 4; i++) acc[i][j] = WMMA_BF16(a[i].v, b.v, acc[i][j]);
    }
  }

  // epilogue: bias + scatter into per-head layouts
#pragma unroll
  for (int i = 0; i < 4; i++) {
#pragma unroll
    for (int j = 0; j < 4; j++) {
      int n = n0 + j * 16 + lr;
      float bv = bias[n];
#pragma unroll
      for (int r = 0; r < 8; r++) {
        int m = m0 + i * 16 + r + hlf * 8;       // token index in [0, 4096)
        float val = acc[i][j][r] + bv;
        int b_ = m >> 11;                        // / 2048
        int s_ = m & 2047;
        if (n < 1024) {
          int h = n >> 6, dd = n & 63;
          Qb[(((size_t)(b_ * 16 + h)) * 2048 + s_) * 64 + dd] = (bf16_t)val;
        } else if (n < 2048) {
          int d = n - 1024; int h = d >> 6, dd = d & 63;
          Kb[(((size_t)(b_ * 16 + h)) * 2048 + s_) * 64 + dd] = (bf16_t)val;
        } else {
          int d = n - 2048; int h = d >> 6, dd = d & 63;
          VTb[(((size_t)(b_ * 16 + h)) * 64 + dd) * 2048 + s_] = (bf16_t)val;
        }
      }
    }
  }
}

// ---------------------------------------------------------------------------
// Flash attention: one wave per 16 queries of one (b,h).
// Per 32-key step: QK^T via 4 WMMAs, causal mask, online softmax (16-lane
// shfl_xor reductions), P staged through LDS (C-layout -> A-layout), then
// O += P * V via 4 WMMAs against V^T.
// ---------------------------------------------------------------------------
__global__ __launch_bounds__(256)
void k_attn(const bf16_t* __restrict__ Qb, const bf16_t* __restrict__ Kb,
            const bf16_t* __restrict__ VTb, bf16_t* __restrict__ Ob) {
  __shared__ __align__(16) bf16_t pstage[8][16][32];  // 8 waves x 1KB

  int w    = threadIdx.x >> 5;
  int wid  = blockIdx.x * 8 + w;
  int lane = threadIdx.x & 31;
  int hlf = lane >> 4, lr = lane & 15;

  const int QBLKS = 2048 / 16;      // 128 query blocks per (b,h)
  int qblk  = wid % QBLKS;
  int bh    = wid / QBLKS;          // 0..31
  int qbase = qblk * 16;

  const bf16_t* Qp  = Qb  + ((size_t)bh * 2048 + qbase) * 64;
  const bf16_t* Kp0 = Kb  + (size_t)bh * 2048 * 64;
  const bf16_t* Vp0 = VTb + (size_t)bh * 64 * 2048;

  // Q fragments (16x64 = two 16x32 A-frags), held in registers for whole loop
  FragAB qf[2];
#pragma unroll
  for (int t = 0; t < 2; t++)
    load_fragA(qf[t], Qp + (size_t)lr * 64 + t * 32 + hlf * 8);

  v8f z = {0.f, 0.f, 0.f, 0.f, 0.f, 0.f, 0.f, 0.f};
  v8f o[4] = {z, z, z, z};
  float m_run[8], l_run[8];
#pragma unroll
  for (int r = 0; r < 8; r++) { m_run[r] = -1e30f; l_run[r] = 0.f; }

  const float scale = 0.125f;  // 1/sqrt(64)

  for (int k0 = 0; k0 < qbase + 16; k0 += 32) {   // causal prune
    v8f s0 = z, s1 = z;
#pragma unroll
    for (int t = 0; t < 2; t++) {
      FragAB kf0, kf1;
      // B operand: lane = key%16, interleaved d-chunks at t*32 + half*8 (+16)
      load_fragB(kf0, Kp0 + (size_t)(k0 + lr) * 64 + t * 32 + hlf * 8);
      load_fragB(kf1, Kp0 + (size_t)(k0 + 16 + lr) * 64 + t * 32 + hlf * 8);
      s0 = WMMA_BF16(qf[t].v, kf0.v, s0);
      s1 = WMMA_BF16(qf[t].v, kf1.v, s1);
    }

    float p0v[8], p1v[8], alpha[8];
#pragma unroll
    for (int r = 0; r < 8; r++) {
      int row = qbase + r + hlf * 8;               // global query index
      float a0 = s0[r] * scale; if (k0 + lr > row)      a0 = -1e30f;
      float a1 = s1[r] * scale; if (k0 + 16 + lr > row) a1 = -1e30f;
      // row-max across the 16 lanes of this half-wave
      float mt = fmaxf(a0, a1);
#pragma unroll
      for (int off = 1; off < 16; off <<= 1) mt = fmaxf(mt, __shfl_xor(mt, off, 32));
      float mnew = fmaxf(m_run[r], mt);
      float al = __expf(m_run[r] - mnew);
      float e0 = __expf(a0 - mnew);
      float e1 = __expf(a1 - mnew);
      float lt = e0 + e1;
#pragma unroll
      for (int off = 1; off < 16; off <<= 1) lt += __shfl_xor(lt, off, 32);
      l_run[r] = l_run[r] * al + lt;
      m_run[r] = mnew;
      alpha[r] = al;
      p0v[r] = e0; p1v[r] = e1;
    }
#pragma unroll
    for (int dt = 0; dt < 4; dt++)
#pragma unroll
      for (int r = 0; r < 8; r++) o[dt][r] *= alpha[r];

    // stage P (C-layout -> LDS row-major 16x32)
#pragma unroll
    for (int r = 0; r < 8; r++) {
      pstage[w][r + hlf * 8][lr]      = (bf16_t)p0v[r];
      pstage[w][r + hlf * 8][16 + lr] = (bf16_t)p1v[r];
    }
    // in-wave DS ordering: stores complete before the A-layout reads below
    asm volatile("s_wait_dscnt 0" ::: "memory");

    FragAB pf;                         // P as A operand (interleaved K map)
    load_fragA(pf, &pstage[w][lr][hlf * 8]);

    // O += P (16x32) * V (32x64); B operand rows of V^T (d-major),
    // interleaved key chunks per lane at k0 + half*8 (+16)
#pragma unroll
    for (int dt = 0; dt < 4; dt++) {
      FragAB vf;
      load_fragB(vf, Vp0 + (size_t)(dt * 16 + lr) * 2048 + k0 + hlf * 8);
      o[dt] = WMMA_BF16(pf.v, vf.v, o[dt]);
    }
  }

  // finalize: divide by l, store bf16 into attn_out [B,S,1024]
  int b_ = bh >> 4, h_ = bh & 15;
#pragma unroll
  for (int r = 0; r < 8; r++) {
    float inv = 1.f / l_run[r];
    int row = qbase + r + hlf * 8;
    size_t base = ((size_t)b_ * 2048 + row) * 1024 + (size_t)h_ * 64;
#pragma unroll
    for (int dt = 0; dt < 4; dt++)
      Ob[base + dt * 16 + lr] = (bf16_t)(o[dt][r] * inv);
  }
}

// ---------------------------------------------------------------------------
// Output projection: attn [4096x1024] bf16 x WprojT [1024x1024] bf16 -> f32
// ---------------------------------------------------------------------------
__global__ __launch_bounds__(256)
void k_gemm_proj(const bf16_t* __restrict__ A, const bf16_t* __restrict__ Bt,
                 const float* __restrict__ bias, float* __restrict__ out) {
  const int NT = 1024 / 64;  // 16
  int wid  = blockIdx.x * 8 + (threadIdx.x >> 5);
  int lane = threadIdx.x & 31;
  int hlf = lane >> 4, lr = lane & 15;
  int m0 = (wid / NT) * 64;
  int n0 = (wid % NT) * 64;

  v8f z = {0.f, 0.f, 0.f, 0.f, 0.f, 0.f, 0.f, 0.f};
  v8f acc[4][4];
#pragma unroll
  for (int i = 0; i < 4; i++)
#pragma unroll
    for (int j = 0; j < 4; j++) acc[i][j] = z;

  for (int k0 = 0; k0 < 1024; k0 += 32) {
    FragAB a[4];
#pragma unroll
    for (int i = 0; i < 4; i++)
      load_fragA(a[i], A + (size_t)(m0 + i * 16 + lr) * 1024 + k0 + hlf * 8);
#pragma unroll
    for (int j = 0; j < 4; j++) {
      FragAB b;
      load_fragB(b, Bt + (size_t)(n0 + j * 16 + lr) * 1024 + k0 + hlf * 8);
#pragma unroll
      for (int i = 0; i < 4; i++) acc[i][j] = WMMA_BF16(a[i].v, b.v, acc[i][j]);
    }
  }

#pragma unroll
  for (int i = 0; i < 4; i++) {
#pragma unroll
    for (int j = 0; j < 4; j++) {
      int n = n0 + j * 16 + lr;
      float bv = bias[n];
#pragma unroll
      for (int r = 0; r < 8; r++) {
        int m = m0 + i * 16 + r + hlf * 8;
        out[(size_t)m * 1024 + n] = acc[i][j][r] + bv;
      }
    }
  }
}

// ---------------------------------------------------------------------------
// Host-side launch
// ---------------------------------------------------------------------------
extern "C" void kernel_launch(void* const* d_in, const int* in_sizes, int n_in,
                              void* d_out, int out_size, void* d_ws, size_t ws_size,
                              hipStream_t stream) {
  const float* x     = (const float*)d_in[0];
  const float* Wqkv  = (const float*)d_in[1];
  const float* bqkv  = (const float*)d_in[2];
  const float* Wproj = (const float*)d_in[3];
  const float* bproj = (const float*)d_in[4];
  float* out = (float*)d_out;

  char* ws = (char*)d_ws;
  bf16_t* xb     = (bf16_t*)ws;  ws += (size_t)4096 * 1024 * 2;  // 8 MB
  bf16_t* WqkvT  = (bf16_t*)ws;  ws += (size_t)3072 * 1024 * 2;  // 6 MB
  bf16_t* WprojT = (bf16_t*)ws;  ws += (size_t)1024 * 1024 * 2;  // 2 MB
  bf16_t* Qb     = (bf16_t*)ws;  ws += (size_t)2 * 16 * 2048 * 64 * 2;  // 8 MB
  bf16_t* Kb     = (bf16_t*)ws;  ws += (size_t)2 * 16 * 2048 * 64 * 2;  // 8 MB
  bf16_t* VTb    = (bf16_t*)ws;  ws += (size_t)2 * 16 * 64 * 2048 * 2;  // 8 MB
  bf16_t* attnb  = (bf16_t*)ws;  ws += (size_t)4096 * 1024 * 2;         // 8 MB

  // 1) precision conversion / weight transposes
  k_cvt<<<(4096 * 1024) / 256, 256, 0, stream>>>(x, xb, 4096 * 1024);
  k_tcvt<<<(1024 * 3072 + 255) / 256, 256, 0, stream>>>(Wqkv, WqkvT, 1024, 3072);
  k_tcvt<<<(1024 * 1024 + 255) / 256, 256, 0, stream>>>(Wproj, WprojT, 1024, 1024);

  // 2) fused QKV projection (WMMA) with per-head scatter
  //    4096/64 * 3072/64 = 3072 wave tiles, 8 waves/block -> 384 blocks
  k_gemm_qkv<<<384, 256, 0, stream>>>(xb, WqkvT, bqkv, Qb, Kb, VTb);

  // 3) causal flash attention: 32 (b,h) * 128 qblocks = 4096 waves -> 512 blocks
  k_attn<<<512, 256, 0, stream>>>(Qb, Kb, VTb, attnb);

  // 4) output projection (WMMA) + bias -> f32 out
  //    4096/64 * 1024/64 = 1024 wave tiles -> 128 blocks
  k_gemm_proj<<<128, 256, 0, stream>>>(attnb, WprojT, bproj, out);
}
